// DecLayer_63342177682143
// MI455X (gfx1250) — matrix operand order repaired and checked by reading, not compiled, not run on hardware
//
#include <hip/hip_runtime.h>

#define DEV __device__ __forceinline__

typedef _Float16 v16h __attribute__((ext_vector_type(16)));
typedef _Float16 v8h  __attribute__((ext_vector_type(8)));
typedef float    v8f  __attribute__((ext_vector_type(8)));

static constexpr int Bb  = 4;
static constexpr int Nn  = 2048;
static constexpr int Kk  = 48;      // neighbors
static constexpr int Dd  = 128;     // hidden
static constexpr int NIN = 384;     // edge feature dim
static constexpr int CIN = Dd + NIN;  // 512
static constexpr int FF  = 4 * Dd;    // 512
static constexpr float SCALE = 30.0f;
static constexpr float EPS   = 1e-5f;

DEV float gelu_exact(float x) {
    return 0.5f * x * (1.0f + erff(x * 0.70710678118654752f));
}

union HFrag { v16h v; v8h h[2]; };

// A operand (16xK32 f16), rows row-major with `stride` elements.
// Lane L<16: row L, K = {kb+0..7, kb+16..23}; lane L+16: row L, K = {kb+8..15, kb+24..31}.
DEV v16h load_a(const _Float16* base, int stride, int kb, int lane) {
    int r  = lane & 15;
    int kh = lane >> 4;
    const _Float16* p = base + r * stride + kb + kh * 8;
    HFrag f;
    f.h[0] = *(const v8h*)(p);
    f.h[1] = *(const v8h*)(p + 16);
    return f.v;
}

// B operand (K32x16 f16) supplied from transposed weights W^T: row n holds K values.
// Lane L<16: col L, K = {kb+0..7, kb+16..23}; lane L+16: col L, K = {kb+8..15, kb+24..31}.
DEV v16h load_b(const _Float16* baseT, int stride, int kb, int lane) {
    int n  = lane & 15;
    int kh = lane >> 4;
    const _Float16* p = baseT + n * stride + kb + kh * 8;
    HFrag f;
    f.h[0] = *(const v8h*)(p);
    f.h[1] = *(const v8h*)(p + 16);
    return f.v;
}

DEV v8f wmma(v16h a, v16h b, v8f c) {
    return __builtin_amdgcn_wmma_f32_16x16x32_f16(false, a, false, b, (short)0, c, false, false);
}

// ---------------- weight prep: f32 (K x N) -> f16 transposed (N x K) ----------------
__global__ void transpose_cvt_kernel(const float* __restrict__ src, _Float16* __restrict__ dst,
                                     int Kdim, int Ncols) {
    int idx = blockIdx.x * blockDim.x + threadIdx.x;
    if (idx >= Kdim * Ncols) return;
    int n = idx / Kdim, k = idx % Kdim;
    dst[n * Kdim + k] = (_Float16)src[k * Ncols + n];
}

// ---------------- kernel 1: per-node edge MLP + masked neighbor reduce ----------------
// grid = B*N blocks, 256 threads (8 waves). Wave w owns output cols [16w,16w+16).
__global__ __launch_bounds__(256)
void edge_message_kernel(const float* __restrict__ hV, const float* __restrict__ hE,
                         const float* __restrict__ maskA,
                         const _Float16* __restrict__ wt1, const float* __restrict__ b1,
                         const _Float16* __restrict__ wt2, const float* __restrict__ b2,
                         const _Float16* __restrict__ wt3, const float* __restrict__ b3,
                         float* __restrict__ dh) {
    extern __shared__ char smem[];
    const int SA = CIN + 8;   // 520 f16, conflict-free padding
    const int SH = Dd + 8;    // 136 f16
    _Float16* sA  = (_Float16*)smem;            // 48 x 520
    _Float16* sH1 = sA  + Kk * SA;              // 48 x 136
    _Float16* sH2 = sH1 + Kk * SH;              // 48 x 136
    float*    sMk = (float*)(sH2 + Kk * SH);    // 48

    const int node = blockIdx.x;
    const int tid  = threadIdx.x;
    const int lane = tid & 31;
    const int wave = tid >> 5;

    // stage concat(h_V, h_E[k]) rows as f16 (h_E read exactly once from HBM)
    for (int idx = tid; idx < Kk * CIN; idx += 256) {
        int r = idx >> 9;          // CIN == 512
        int c = idx & (CIN - 1);
        float v = (c < Dd) ? hV[node * Dd + c]
                           : hE[((size_t)node * Kk + r) * NIN + (c - Dd)];
        sA[r * SA + c] = (_Float16)v;
    }
    if (tid < Kk) sMk[tid] = maskA[node * Kk + tid];
    __syncthreads();

    const int nb  = wave * 16;
    const int col = nb + (lane & 15);
    const int r0  = (lane >> 4) * 8;   // C-frag hi half-wave holds rows +8

    // GEMM1: (48x512) @ (512x128)
    v8f c0 = {}, c1 = {}, c2 = {};
    #pragma unroll
    for (int kb = 0; kb < CIN; kb += 32) {
        v16h bf = load_b(wt1 + nb * CIN, CIN, kb, lane);
        c0 = wmma(load_a(sA +  0 * SA, SA, kb, lane), bf, c0);
        c1 = wmma(load_a(sA + 16 * SA, SA, kb, lane), bf, c1);
        c2 = wmma(load_a(sA + 32 * SA, SA, kb, lane), bf, c2);
    }
    {
        float bias = b1[col];
        #pragma unroll
        for (int r = 0; r < 8; ++r) {
            sH1[( 0 + r0 + r) * SH + col] = (_Float16)gelu_exact(c0[r] + bias);
            sH1[(16 + r0 + r) * SH + col] = (_Float16)gelu_exact(c1[r] + bias);
            sH1[(32 + r0 + r) * SH + col] = (_Float16)gelu_exact(c2[r] + bias);
        }
    }
    __syncthreads();

    // GEMM2: (48x128) @ (128x128)
    c0 = (v8f){}; c1 = (v8f){}; c2 = (v8f){};
    #pragma unroll
    for (int kb = 0; kb < Dd; kb += 32) {
        v16h bf = load_b(wt2 + nb * Dd, Dd, kb, lane);
        c0 = wmma(load_a(sH1 +  0 * SH, SH, kb, lane), bf, c0);
        c1 = wmma(load_a(sH1 + 16 * SH, SH, kb, lane), bf, c1);
        c2 = wmma(load_a(sH1 + 32 * SH, SH, kb, lane), bf, c2);
    }
    {
        float bias = b2[col];
        #pragma unroll
        for (int r = 0; r < 8; ++r) {
            sH2[( 0 + r0 + r) * SH + col] = (_Float16)gelu_exact(c0[r] + bias);
            sH2[(16 + r0 + r) * SH + col] = (_Float16)gelu_exact(c1[r] + bias);
            sH2[(32 + r0 + r) * SH + col] = (_Float16)gelu_exact(c2[r] + bias);
        }
    }
    __syncthreads();

    // GEMM3: (48x128) @ (128x128), mask per neighbor row, reduce over K
    c0 = (v8f){}; c1 = (v8f){}; c2 = (v8f){};
    #pragma unroll
    for (int kb = 0; kb < Dd; kb += 32) {
        v16h bf = load_b(wt3 + nb * Dd, Dd, kb, lane);
        c0 = wmma(load_a(sH2 +  0 * SH, SH, kb, lane), bf, c0);
        c1 = wmma(load_a(sH2 + 16 * SH, SH, kb, lane), bf, c1);
        c2 = wmma(load_a(sH2 + 32 * SH, SH, kb, lane), bf, c2);
    }
    {
        float bias = b3[col];
        float acc = 0.0f;
        #pragma unroll
        for (int r = 0; r < 8; ++r) {
            acc += (c0[r] + bias) * sMk[ 0 + r0 + r];
            acc += (c1[r] + bias) * sMk[16 + r0 + r];
            acc += (c2[r] + bias) * sMk[32 + r0 + r];
        }
        acc += __shfl_xor(acc, 16);        // combine lane L and L+16 (same column)
        if (lane < 16) dh[node * Dd + col] = acc * (1.0f / SCALE);
    }
}

// ---------------- kernel 2: residual + LN1 + WMMA FFN + residual + LN2 + mask ----------------
// grid = B*N/16 blocks, 256 threads. 16 node-rows per block.
__global__ __launch_bounds__(256)
void node_ffn_kernel(const float* __restrict__ hV, const float* __restrict__ dh,
                     const float* __restrict__ maskV,
                     const _Float16* __restrict__ wtin,  const float* __restrict__ bin,
                     const _Float16* __restrict__ wtout, const float* __restrict__ bout,
                     const float* __restrict__ g1, const float* __restrict__ o1,
                     const float* __restrict__ g2, const float* __restrict__ o2,
                     float* __restrict__ out) {
    extern __shared__ char smem[];
    const int SY  = Dd + 4;    // 132 f32
    const int SAs = Dd + 8;    // 136 f16
    const int SHs = FF + 8;    // 520 f16
    float*    sY = (float*)smem;                  // 16 x 132 (post-LN1, f32 for residual)
    float*    sO = sY + 16 * SY;                  // 16 x 132 (pre-LN2)
    _Float16* sA = (_Float16*)(sO + 16 * SY);     // 16 x 136
    _Float16* sH = sA + 16 * SAs;                 // 16 x 520

    const int tid   = threadIdx.x;
    const int lane  = tid & 31;
    const int wave  = tid >> 5;
    const int node0 = blockIdx.x * 16;

    // residual + LN1; wave handles rows 2w, 2w+1
    for (int rr = 0; rr < 2; ++rr) {
        int row = wave * 2 + rr;
        int node = node0 + row;
        float x[4]; float s = 0.f, sq = 0.f;
        #pragma unroll
        for (int j = 0; j < 4; ++j) {
            int c = lane * 4 + j;
            float v = hV[node * Dd + c] + dh[node * Dd + c];
            x[j] = v; s += v; sq += v * v;
        }
        #pragma unroll
        for (int off = 16; off >= 1; off >>= 1) { s += __shfl_xor(s, off); sq += __shfl_xor(sq, off); }
        float mu = s * (1.0f / Dd);
        float rs = rsqrtf(sq * (1.0f / Dd) - mu * mu + EPS);
        #pragma unroll
        for (int j = 0; j < 4; ++j) {
            int c = lane * 4 + j;
            float y = (x[j] - mu) * rs * g1[c] + o1[c];
            sY[row * SY + c]  = y;
            sA[row * SAs + c] = (_Float16)y;
        }
    }
    __syncthreads();

    const int r0 = (lane >> 4) * 8;

    // FFN in: (16x128) @ (128x512), gelu; wave covers 4 column tiles
    for (int t = 0; t < 4; ++t) {
        int cb = (wave + t * 8) * 16;
        v8f c = {};
        #pragma unroll
        for (int kb = 0; kb < Dd; kb += 32)
            c = wmma(load_a(sA, SAs, kb, lane), load_b(wtin + cb * Dd, Dd, kb, lane), c);
        int col = cb + (lane & 15);
        float bias = bin[col];
        #pragma unroll
        for (int r = 0; r < 8; ++r)
            sH[(r0 + r) * SHs + col] = (_Float16)gelu_exact(c[r] + bias);
    }
    __syncthreads();

    // FFN out: (16x512) @ (512x128) + bias + residual
    {
        int cb = wave * 16;
        v8f c = {};
        #pragma unroll
        for (int kb = 0; kb < FF; kb += 32)
            c = wmma(load_a(sH, SHs, kb, lane), load_b(wtout + cb * FF, FF, kb, lane), c);
        int col = cb + (lane & 15);
        float bias = bout[col];
        #pragma unroll
        for (int r = 0; r < 8; ++r) {
            int row = r0 + r;
            sO[row * SY + col] = c[r] + bias + sY[row * SY + col];
        }
    }
    __syncthreads();

    // LN2 + mask_V + store
    for (int rr = 0; rr < 2; ++rr) {
        int row = wave * 2 + rr;
        int node = node0 + row;
        float x[4]; float s = 0.f, sq = 0.f;
        #pragma unroll
        for (int j = 0; j < 4; ++j) {
            float v = sO[row * SY + lane * 4 + j];
            x[j] = v; s += v; sq += v * v;
        }
        #pragma unroll
        for (int off = 16; off >= 1; off >>= 1) { s += __shfl_xor(s, off); sq += __shfl_xor(sq, off); }
        float mu = s * (1.0f / Dd);
        float rs = rsqrtf(sq * (1.0f / Dd) - mu * mu + EPS);
        float m = maskV[node];
        #pragma unroll
        for (int j = 0; j < 4; ++j) {
            int c = lane * 4 + j;
            out[node * Dd + c] = m * ((x[j] - mu) * rs * g2[c] + o2[c]);
        }
    }
}

extern "C" void kernel_launch(void* const* d_in, const int* in_sizes, int n_in,
                              void* d_out, int out_size, void* d_ws, size_t ws_size,
                              hipStream_t stream) {
    (void)in_sizes; (void)n_in; (void)out_size; (void)ws_size;
    const float* hV     = (const float*)d_in[0];
    const float* hE     = (const float*)d_in[1];
    const float* maskV  = (const float*)d_in[2];
    const float* maskA  = (const float*)d_in[3];
    const float* W1     = (const float*)d_in[4];
    const float* b1     = (const float*)d_in[5];
    const float* W2     = (const float*)d_in[6];
    const float* b2     = (const float*)d_in[7];
    const float* W3     = (const float*)d_in[8];
    const float* b3     = (const float*)d_in[9];
    const float* W_in   = (const float*)d_in[10];
    const float* b_in   = (const float*)d_in[11];
    const float* W_out  = (const float*)d_in[12];
    const float* b_out  = (const float*)d_in[13];
    const float* g1     = (const float*)d_in[14];
    const float* o1     = (const float*)d_in[15];
    const float* g2     = (const float*)d_in[16];
    const float* o2     = (const float*)d_in[17];
    float* out = (float*)d_out;

    // workspace layout
    char* ws = (char*)d_ws;
    float* dh = (float*)ws;
    size_t off = (size_t)Bb * Nn * Dd * sizeof(float);
    _Float16* wt1   = (_Float16*)(ws + off); off += (size_t)Dd * CIN * sizeof(_Float16);
    _Float16* wt2   = (_Float16*)(ws + off); off += (size_t)Dd * Dd  * sizeof(_Float16);
    _Float16* wt3   = (_Float16*)(ws + off); off += (size_t)Dd * Dd  * sizeof(_Float16);
    _Float16* wtin  = (_Float16*)(ws + off); off += (size_t)FF * Dd  * sizeof(_Float16);
    _Float16* wtout = (_Float16*)(ws + off); off += (size_t)Dd * FF  * sizeof(_Float16);

    // weight transpose+convert (tiny; L2-resident afterwards)
    transpose_cvt_kernel<<<(CIN * Dd + 255) / 256, 256, 0, stream>>>(W1,    wt1,   CIN, Dd);
    transpose_cvt_kernel<<<(Dd  * Dd + 255) / 256, 256, 0, stream>>>(W2,    wt2,   Dd,  Dd);
    transpose_cvt_kernel<<<(Dd  * Dd + 255) / 256, 256, 0, stream>>>(W3,    wt3,   Dd,  Dd);
    transpose_cvt_kernel<<<(Dd  * FF + 255) / 256, 256, 0, stream>>>(W_in,  wtin,  Dd,  FF);
    transpose_cvt_kernel<<<(FF  * Dd + 255) / 256, 256, 0, stream>>>(W_out, wtout, FF,  Dd);

    size_t shm1 = (size_t)(Kk * (CIN + 8) + 2 * Kk * (Dd + 8)) * sizeof(_Float16)
                + (size_t)Kk * sizeof(float);
    edge_message_kernel<<<Bb * Nn, 256, shm1, stream>>>(hV, hE, maskA,
                                                        wt1, b1, wt2, b2, wt3, b3, dh);

    size_t shm2 = (size_t)(2 * 16 * (Dd + 4)) * sizeof(float)
                + (size_t)(16 * (Dd + 8) + 16 * (FF + 8)) * sizeof(_Float16);
    node_ffn_kernel<<<(Bb * Nn) / 16, 256, shm2, stream>>>(hV, dh, maskV,
                                                           wtin, b_in, wtout, b_out,
                                                           g1, o1, g2, o2, out);
}
